// Decoder_26792005992845
// MI455X (gfx1250) — compile-verified, matrix operand order checked by reading, not run on hardware
//
#include <hip/hip_runtime.h>

// ---------------- CDNA5 WMMA types ----------------
typedef __attribute__((ext_vector_type(16))) __bf16        v16bf;
typedef __attribute__((ext_vector_type(8)))  float         v8f;
typedef __attribute__((ext_vector_type(4)))  unsigned int  v4u;
typedef __attribute__((ext_vector_type(4)))  float         v4f;
typedef unsigned short u16;

union FragU { v16bf v; v4u q[2]; u16 u[16]; };

__device__ __forceinline__ u16 f2bf(float f) {
    unsigned x = __float_as_uint(f);
    x += 0x7FFFu + ((x >> 16) & 1u);        // round-to-nearest-even
    return (u16)(x >> 16);
}
__device__ __forceinline__ float bf2f(u16 v) {
    return __uint_as_float((unsigned)v << 16);
}
__device__ __forceinline__ float sigf(float x) { return 1.f / (1.f + __expf(-x)); }

// ---------------- problem dims ----------------
#define Bn      64
#define Ln      196
#define ENCn    2048
#define DECn    512
#define EMBn    512
#define ATTn    512
#define Vn      20000
#define Tn      26
#define TSTEPS  25                  // T-1 decode steps
#define XK      (EMBn + ENCn + DECn)   // 3072: fused LSTM input [emb|gate*z|h]

// =====================================================================
// Wave-per-N-tile WMMA GEMM, MREP M-tiles per wave, double-buffered K.
//   C[M,N] = act(A[M,K] * Bt^T + bias),  Bt stored [N,K] (K contiguous).
//   One B fragment feeds MREP WMMAs (weights are the streamed operand;
//   M=64 is tiny), and fragment loads for K-step k+32 are issued before
//   the WMMAs of step k (ping-pong), so vmem latency is hidden.
//   Fragments: two contiguous 16B loads per lane (CDNA5 16-bit layout).
//   K must be a multiple of 64.
// =====================================================================
template <int ACT, bool AF32, int MREP>
__global__ __launch_bounds__(256) void gemm16(
    const u16* __restrict__ Abf, const float* __restrict__ Af,
    const int* __restrict__ rowidx, const u16* __restrict__ Bt,
    const float* __restrict__ bias, float* __restrict__ C,
    u16* __restrict__ Cbf, int Nt, int K, int lda, long ldc,
    const int* __restrict__ lensMask, int t)
{
    const int wave  = threadIdx.x >> 5;
    const int lane  = threadIdx.x & 31;
    const int ntile = blockIdx.x * (blockDim.x >> 5) + wave;
    if (ntile >= Nt) return;                       // wave-uniform
    const int mb    = blockIdx.y * MREP;           // first M-tile
    const int half  = lane >> 4;                   // K-half selector
    const int lr    = lane & 15;                   // row (A) / col (B)

    const u16* bbase = Bt + ((long)ntile * 16 + lr) * (long)K;

    long asrc[MREP];
#pragma unroll
    for (int r = 0; r < MREP; r++) {
        const int arow = (mb + r) * 16 + lr;
        asrc[r] = rowidx ? (long)rowidx[arow] : (long)arow;
    }

    v8f acc[MREP];
#pragma unroll
    for (int r = 0; r < MREP; r++) acc[r] = (v8f){0.f,0.f,0.f,0.f,0.f,0.f,0.f,0.f};

    auto load_group = [&](FragU& b, FragU (&a)[MREP], int kk) {
        const int k0 = kk + half * 8;
        const v4u* bp = (const v4u*)(bbase + k0);
        b.q[0] = bp[0];                            // K = k0 .. k0+7
        b.q[1] = bp[2];                            // K = k0+16 .. k0+23
#pragma unroll
        for (int r = 0; r < MREP; r++) {
            if constexpr (AF32) {
                const v4f* ap = (const v4f*)(Af + asrc[r] * (long)lda + k0);
                const v4f f0 = ap[0], f1 = ap[1];  // K = k0..k0+7
                const v4f f2 = ap[4], f3 = ap[5];  // K = k0+16..k0+23
#pragma unroll
                for (int e = 0; e < 4; e++) {
                    a[r].u[e]      = f2bf(f0[e]);
                    a[r].u[4 + e]  = f2bf(f1[e]);
                    a[r].u[8 + e]  = f2bf(f2[e]);
                    a[r].u[12 + e] = f2bf(f3[e]);
                }
            } else {
                const v4u* ap = (const v4u*)(Abf + asrc[r] * (long)lda + k0);
                a[r].q[0] = ap[0];
                a[r].q[1] = ap[2];
            }
        }
    };

    FragU b0, b1;
    FragU a0[MREP], a1[MREP];
    load_group(b0, a0, 0);
    for (int k = 0; k < K; k += 64) {
        load_group(b1, a1, k + 32);                       // next half-step
        __builtin_prefetch(bbase + k + half * 8 + 128, 0, 1); // global_prefetch_b8
#pragma unroll
        for (int r = 0; r < MREP; r++)
            acc[r] = __builtin_amdgcn_wmma_f32_16x16x32_bf16(
                         false, a0[r].v, false, b0.v, (short)0, acc[r], false, false);
        if (k + 64 < K) load_group(b0, a0, k + 64);
#pragma unroll
        for (int r = 0; r < MREP; r++)
            acc[r] = __builtin_amdgcn_wmma_f32_16x16x32_bf16(
                         false, a1[r].v, false, b1.v, (short)0, acc[r], false, false);
    }

    const int   gn = ntile * 16 + lr;
    const float bv = bias ? bias[gn] : 0.f;
#pragma unroll
    for (int r = 0; r < MREP; r++) {
#pragma unroll
        for (int j = 0; j < 8; j++) {
            const int gm = (mb + r) * 16 + j + half * 8;
            float v = acc[r][j] + bv;
            if (ACT == 1) v = tanhf(v);
            if (ACT == 2) v = sigf(v);
            const long ci = (long)gm * ldc + gn;
            if (lensMask && t >= lensMask[gm]) v = 0.f;
            if (C)   C[ci]   = v;
            if (Cbf) Cbf[ci] = f2bf(v);
        }
    }
}

// ---------------- setup kernels ----------------
__global__ void sort_setup(const int* __restrict__ lens_in, const int* __restrict__ cap,
                           int* __restrict__ sort_ind, int* __restrict__ lens_sorted,
                           int* __restrict__ cap_sorted, float* __restrict__ out_cap,
                           float* __restrict__ out_lens)
{
    __shared__ int L[Bn];
    const int i = threadIdx.x;
    L[i] = lens_in[i];
    __syncthreads();
    const int li = L[i];
    int r = 0;
    for (int j = 0; j < Bn; j++) {
        const int lj = L[j];
        r += (lj > li) || (lj == li && j < i);     // stable desc sort rank
    }
    sort_ind[r]    = i;
    lens_sorted[r] = li - 1;
    out_lens[r]    = (float)(li - 1);
    for (int k = 0; k < Tn; k++) {
        const int tok = cap[i * Tn + k];
        cap_sorted[r * Tn + k] = tok;
        out_cap[r * Tn + k]    = (float)tok;
    }
}

__global__ void build_rowidx(const int* __restrict__ sort_ind, int* __restrict__ rowidx)
{
    int r = blockIdx.x * blockDim.x + threadIdx.x;
    if (r < Bn * Ln) rowidx[r] = sort_ind[r / Ln] * Ln + (r % Ln);
}

__global__ void cvt_copy(u16* __restrict__ dst, const float* __restrict__ src, long n)
{
    long i      = (long)blockIdx.x * blockDim.x + threadIdx.x;
    long stride = (long)gridDim.x * blockDim.x;
    for (; i < n; i += stride) dst[i] = f2bf(src[i]);
}

// copy src[rows,cols] (f32) into dst rows of width dld at column doff (bf16)
__global__ void cvt_copy_strided(u16* __restrict__ dst, const float* __restrict__ src,
                                 long rows, long cols, long dld, long doff)
{
    long n      = rows * cols;
    long i      = (long)blockIdx.x * blockDim.x + threadIdx.x;
    long stride = (long)gridDim.x * blockDim.x;
    for (; i < n; i += stride) {
        long r = i / cols, c = i % cols;
        dst[r * dld + doff + c] = f2bf(src[i]);
    }
}

// src: R x Cc (row-major) -> dst: Cc x R (row-major, transpose), bf16
__global__ void cvt_transpose(u16* __restrict__ dst, const float* __restrict__ src,
                              int R, int Cc)
{
    long n      = (long)R * Cc;
    long i      = (long)blockIdx.x * blockDim.x + threadIdx.x;
    long stride = (long)gridDim.x * blockDim.x;
    for (; i < n; i += stride) {
        long r = i % R, c = i / R;
        dst[i] = f2bf(src[r * (long)Cc + c]);
    }
}

// enc_bf[r, :] = bf16(enc[rowidx[r], :])   (sorted bf16 encoder copy)
__global__ void cvt_gather_rows(u16* __restrict__ dst, const float* __restrict__ src,
                                const int* __restrict__ rowidx, long rows, long cols)
{
    long n      = rows * cols;
    long i      = (long)blockIdx.x * blockDim.x + threadIdx.x;
    long stride = (long)gridDim.x * blockDim.x;
    for (; i < n; i += stride) {
        long r = i / cols, c = i % cols;
        dst[i] = f2bf(src[(long)rowidx[r] * cols + c]);
    }
}

__global__ void add_bias2(float* __restrict__ dst, const float* __restrict__ a,
                          const float* __restrict__ b, int n)
{
    int i = blockIdx.x * blockDim.x + threadIdx.x;
    if (i < n) dst[i] = a[i] + b[i];
}

__global__ void embed_gather(const float* __restrict__ embW,
                             const int* __restrict__ cap_sorted,
                             u16* __restrict__ emb_bf)
{
    int i = blockIdx.x * blockDim.x + threadIdx.x;
    if (i >= Bn * TSTEPS * EMBn) return;
    int e = i % EMBn, t = (i / EMBn) % TSTEPS, b = i / (EMBn * TSTEPS);
    int tok = cap_sorted[b * Tn + t];
    emb_bf[i] = f2bf(embW[(long)tok * EMBn + e]);
}

template <bool BF>
__global__ void mean_enc(const float* __restrict__ encf, const u16* __restrict__ encb,
                         const int* __restrict__ sort_ind, u16* __restrict__ mean_bf)
{
    int i = blockIdx.x * blockDim.x + threadIdx.x;
    if (i >= Bn * ENCn) return;
    int e = i % ENCn, b = i / ENCn;
    float s = 0.f;
    if constexpr (BF) {
        const u16* p = encb + (long)b * Ln * ENCn + e;      // already sorted
        for (int l = 0; l < Ln; l++) s += bf2f(p[(long)l * ENCn]);
    } else {
        const float* p = encf + (long)sort_ind[b] * Ln * ENCn + e;
        for (int l = 0; l < Ln; l++) s += p[(long)l * ENCn];
    }
    mean_bf[i] = f2bf(s * (1.f / Ln));
}

// ---------------- per-step kernels ----------------
// one wave32 per (b,l): scores[b,l] = sum_a tanh(enc_att+dec_att)*Wa + ba
__global__ void attn_scores(const float* __restrict__ enc_att,
                            const float* __restrict__ dec_att,
                            const float* __restrict__ Wa, const float* __restrict__ ba,
                            float* __restrict__ scores)
{
    int g    = (blockIdx.x * blockDim.x + threadIdx.x) >> 5;
    int lane = threadIdx.x & 31;
    if (g >= Bn * Ln) return;
    int b = g / Ln;
    const float* u = enc_att + (long)g * ATTn;
    const float* d = dec_att + (long)b * ATTn;
    float s = 0.f;
    for (int a = lane; a < ATTn; a += 32) s += tanhf(u[a] + d[a]) * Wa[a];
    for (int off = 16; off > 0; off >>= 1) s += __shfl_xor(s, off, 32);
    if (lane == 0) scores[g] = s + ba[0];
}

__global__ __launch_bounds__(256) void softmax_alpha(
    const float* __restrict__ scores, float* __restrict__ alpha,
    float* __restrict__ alphas_out, const int* __restrict__ lens, int t)
{
    __shared__ float red[256];
    const int b = blockIdx.x, tid = threadIdx.x;
    float v = (tid < Ln) ? scores[b * Ln + tid] : -3.4e38f;
    red[tid] = v; __syncthreads();
    for (int o = 128; o > 0; o >>= 1) {
        if (tid < o) red[tid] = fmaxf(red[tid], red[tid + o]);
        __syncthreads();
    }
    const float mx = red[0]; __syncthreads();
    float e = (tid < Ln) ? __expf(v - mx) : 0.f;
    red[tid] = e; __syncthreads();
    for (int o = 128; o > 0; o >>= 1) {
        if (tid < o) red[tid] += red[tid + o];
        __syncthreads();
    }
    const float inv = 1.f / red[0];
    if (tid < Ln) {
        float a = e * inv;
        alpha[b * Ln + tid] = a;
        float m = (t < lens[b]) ? 1.f : 0.f;
        alphas_out[((long)b * TSTEPS + t) * Ln + tid] = a * m;
    }
}

// x = [emb_t | gate * (alpha^T enc) | h]  in bf16 -> fused LSTM GEMM A
template <bool BF>
__global__ void context_build_x(const float* __restrict__ encf, const u16* __restrict__ encb,
                                const int* __restrict__ sort_ind,
                                const float* __restrict__ alpha,
                                const float* __restrict__ gate,
                                const u16* __restrict__ emb_bf,
                                const u16* __restrict__ h_bf,
                                u16* __restrict__ x_bf, int t)
{
    int i = blockIdx.x * blockDim.x + threadIdx.x;
    if (i >= Bn * XK) return;
    int j = i % XK, b = i / XK;
    if (j < EMBn) {
        x_bf[i] = emb_bf[((long)b * TSTEPS + t) * EMBn + j];
    } else if (j < EMBn + ENCn) {
        int e = j - EMBn;
        const float* al = alpha + b * Ln;
        float s = 0.f;
        if constexpr (BF) {
            const u16* p = encb + (long)b * Ln * ENCn + e;  // already sorted
            for (int l = 0; l < Ln; l++) s += al[l] * bf2f(p[(long)l * ENCn]);
        } else {
            const float* p = encf + (long)sort_ind[b] * Ln * ENCn + e;
            for (int l = 0; l < Ln; l++) s += al[l] * p[(long)l * ENCn];
        }
        x_bf[i] = f2bf(gate[(long)b * ENCn + e] * s);
    } else {
        x_bf[i] = h_bf[(long)b * DECn + (j - EMBn - ENCn)];
    }
}

__global__ void lstm_cell(const float* __restrict__ g, float* __restrict__ c,
                          u16* __restrict__ h_bf)
{
    int i = blockIdx.x * blockDim.x + threadIdx.x;
    if (i >= Bn * DECn) return;
    int d = i % DECn, b = i / DECn;
    const float* gr = g + (long)b * 4 * DECn;
    float ii = sigf(gr[d]);
    float ff = sigf(gr[DECn + d]);
    float gg = tanhf(gr[2 * DECn + d]);
    float oo = sigf(gr[3 * DECn + d]);
    float cn = ff * c[i] + ii * gg;
    c[i]    = cn;
    h_bf[i] = f2bf(oo * tanhf(cn));
}

// =====================================================================
extern "C" void kernel_launch(void* const* d_in, const int* in_sizes, int n_in,
                              void* d_out, int out_size, void* d_ws, size_t ws_size,
                              hipStream_t stream)
{
    const float* enc  = (const float*)d_in[0];
    const int*   cap  = (const int*)  d_in[1];
    const int*   clen = (const int*)  d_in[2];
    const float* embW = (const float*)d_in[3];
    const float* We   = (const float*)d_in[4];
    const float* be   = (const float*)d_in[5];
    const float* Wd   = (const float*)d_in[6];
    const float* bd   = (const float*)d_in[7];
    const float* Wa   = (const float*)d_in[8];
    const float* ba   = (const float*)d_in[9];
    const float* Wih  = (const float*)d_in[10];
    const float* bih  = (const float*)d_in[11];
    const float* Whh  = (const float*)d_in[12];
    const float* bhh  = (const float*)d_in[13];
    const float* Wfb  = (const float*)d_in[14];
    const float* bfb  = (const float*)d_in[15];
    const float* Wh0  = (const float*)d_in[16];
    const float* bh0  = (const float*)d_in[17];
    const float* Wc0  = (const float*)d_in[18];
    const float* bc0  = (const float*)d_in[19];
    const float* Wfc  = (const float*)d_in[20];
    const float* bfc  = (const float*)d_in[21];
    (void)in_sizes; (void)n_in; (void)out_size;

    float* preds      = (float*)d_out;                               // [B,25,V]
    float* alphas_out = preds + (long)Bn * TSTEPS * Vn;              // [B,25,L]
    float* cap_out    = alphas_out + (long)Bn * TSTEPS * Ln;         // [B,26]
    float* lens_out   = cap_out + Bn * Tn;                           // [B]

    char* w = (char*)d_ws;
    auto alloc = [&](size_t bytes) -> void* {
        void* p = (void*)w;
        w += (bytes + 255) & ~(size_t)255;
        return p;
    };
    int* sort_ind    = (int*)alloc(Bn * 4);
    int* lens_sorted = (int*)alloc(Bn * 4);
    int* cap_sorted  = (int*)alloc((size_t)Bn * Tn * 4);
    int* rowidx      = (int*)alloc((size_t)Bn * Ln * 4);
    u16* WeT   = (u16*)alloc((size_t)ATTn * ENCn * 2);               // [ATT,ENC]
    u16* WdT   = (u16*)alloc((size_t)ATTn * DECn * 2);               // [ATT,DEC]
    u16* WfbT  = (u16*)alloc((size_t)ENCn * DECn * 2);               // [ENC,DEC]
    u16* Wh0T  = (u16*)alloc((size_t)DECn * ENCn * 2);               // [DEC,ENC]
    u16* Wc0T  = (u16*)alloc((size_t)DECn * ENCn * 2);               // [DEC,ENC]
    u16* WfcT  = (u16*)alloc((size_t)Vn * DECn * 2);                 // [V,DEC]
    u16* WihhB = (u16*)alloc((size_t)4 * DECn * XK * 2);             // [4*DEC, XK]
    float* bih_comb = (float*)alloc((size_t)4 * DECn * 4);
    u16* emb_bf  = (u16*)alloc((size_t)Bn * TSTEPS * EMBn * 2);
    u16* mean_bf = (u16*)alloc((size_t)Bn * ENCn * 2);
    float* enc_att = (float*)alloc((size_t)Bn * Ln * ATTn * 4);
    float* dec_att = (float*)alloc((size_t)Bn * ATTn * 4);
    float* scores  = (float*)alloc((size_t)Bn * Ln * 4);
    float* alpha   = (float*)alloc((size_t)Bn * Ln * 4);
    float* gatev   = (float*)alloc((size_t)Bn * ENCn * 4);
    u16*   x_bf    = (u16*)alloc((size_t)Bn * XK * 2);
    float* gbuf    = (float*)alloc((size_t)Bn * 4 * DECn * 4);
    u16*   h_bf    = (u16*)alloc((size_t)Bn * DECn * 2);
    float* cst     = (float*)alloc((size_t)Bn * DECn * 4);

    // Optional: sorted bf16 encoder copy (pure-bf16 big GEMM + halved
    // per-step context traffic) if workspace allows. ws_size is fixed per
    // run, so this choice is deterministic.
    const size_t encbf_bytes = (size_t)Bn * Ln * ENCn * 2;
    size_t used = (size_t)(w - (char*)d_ws);
    u16* enc_bf = nullptr;
    if (ws_size >= used + encbf_bytes + 256) enc_bf = (u16*)alloc(encbf_bytes);

    // ---- setup ----
    sort_setup<<<1, Bn, 0, stream>>>(clen, cap, sort_ind, lens_sorted, cap_sorted,
                                     cap_out, lens_out);
    build_rowidx<<<(Bn * Ln + 255) / 256, 256, 0, stream>>>(sort_ind, rowidx);
    cvt_transpose<<<512, 256, 0, stream>>>(WeT,  We,  ENCn, ATTn);
    cvt_transpose<<<256, 256, 0, stream>>>(WdT,  Wd,  DECn, ATTn);
    cvt_transpose<<<512, 256, 0, stream>>>(WfbT, Wfb, DECn, ENCn);
    cvt_transpose<<<512, 256, 0, stream>>>(Wh0T, Wh0, ENCn, DECn);
    cvt_transpose<<<512, 256, 0, stream>>>(Wc0T, Wc0, ENCn, DECn);
    cvt_transpose<<<1024, 256, 0, stream>>>(WfcT, Wfc, DECn, Vn);
    // fused LSTM weight: rows [4*DEC], cols [Wih(2560) | Whh(512)]
    cvt_copy_strided<<<1024, 256, 0, stream>>>(WihhB, Wih, 4 * DECn, ENCn + EMBn, XK, 0);
    cvt_copy_strided<<<512, 256, 0, stream>>>(WihhB, Whh, 4 * DECn, DECn, XK, ENCn + EMBn);
    add_bias2<<<(4 * DECn + 255) / 256, 256, 0, stream>>>(bih_comb, bih, bhh, 4 * DECn);
    embed_gather<<<(Bn * TSTEPS * EMBn + 255) / 256, 256, 0, stream>>>(embW, cap_sorted, emb_bf);

    if (enc_bf) {
        cvt_gather_rows<<<2048, 256, 0, stream>>>(enc_bf, enc, rowidx, (long)Bn * Ln, ENCn);
        mean_enc<true><<<(Bn * ENCn + 255) / 256, 256, 0, stream>>>(nullptr, enc_bf, sort_ind, mean_bf);
        // enc_att = enc_sorted @ We + be  (pure bf16 path)
        gemm16<0, false, 4><<<dim3(4, 196), 256, 0, stream>>>(
            enc_bf, nullptr, nullptr, WeT, be, enc_att, nullptr,
            ATTn / 16, ENCn, ENCn, (long)ATTn, nullptr, 0);
    } else {
        mean_enc<false><<<(Bn * ENCn + 255) / 256, 256, 0, stream>>>(enc, nullptr, sort_ind, mean_bf);
        gemm16<0, true, 4><<<dim3(4, 196), 256, 0, stream>>>(
            nullptr, enc, rowidx, WeT, be, enc_att, nullptr,
            ATTn / 16, ENCn, ENCn, (long)ATTn, nullptr, 0);
    }
    // h0 = tanh(mean @ Wh0 + bh0) -> bf16 h ;  c0 = tanh(mean @ Wc0 + bc0) -> f32 c
    gemm16<1, false, 4><<<dim3(8, 1), 128, 0, stream>>>(
        mean_bf, nullptr, nullptr, Wh0T, bh0, nullptr, h_bf,
        DECn / 16, ENCn, ENCn, (long)DECn, nullptr, 0);
    gemm16<1, false, 4><<<dim3(8, 1), 128, 0, stream>>>(
        mean_bf, nullptr, nullptr, Wc0T, bc0, cst, nullptr,
        DECn / 16, ENCn, ENCn, (long)DECn, nullptr, 0);

    // ---- 25 sequential decode steps ----
    for (int t = 0; t < TSTEPS; t++) {
        // dec_att = h @ Wd + bd
        gemm16<0, false, 4><<<dim3(8, 1), 128, 0, stream>>>(
            h_bf, nullptr, nullptr, WdT, bd, dec_att, nullptr,
            ATTn / 16, DECn, DECn, (long)ATTn, nullptr, 0);
        attn_scores<<<(Bn * Ln * 32 + 255) / 256, 256, 0, stream>>>(
            enc_att, dec_att, Wa, ba, scores);
        softmax_alpha<<<Bn, 256, 0, stream>>>(scores, alpha, alphas_out, lens_sorted, t);
        // gate = sigmoid(h @ Wfb + bfb)
        gemm16<2, false, 4><<<dim3(32, 1), 128, 0, stream>>>(
            h_bf, nullptr, nullptr, WfbT, bfb, gatev, nullptr,
            ENCn / 16, DECn, DECn, (long)ENCn, nullptr, 0);
        // x = [emb_t | gate * (alpha^T enc) | h]
        if (enc_bf)
            context_build_x<true><<<(Bn * XK + 255) / 256, 256, 0, stream>>>(
                nullptr, enc_bf, sort_ind, alpha, gatev, emb_bf, h_bf, x_bf, t);
        else
            context_build_x<false><<<(Bn * XK + 255) / 256, 256, 0, stream>>>(
                enc, nullptr, sort_ind, alpha, gatev, emb_bf, h_bf, x_bf, t);
        // g = x @ [Wih|Whh]^T + (bih+bhh)   (fused LSTM GEMM, K=3072)
        gemm16<0, false, 4><<<dim3(32, 1), 128, 0, stream>>>(
            x_bf, nullptr, nullptr, WihhB, bih_comb, gbuf, nullptr,
            4 * DECn / 16, XK, XK, (long)(4 * DECn), nullptr, 0);
        lstm_cell<<<(Bn * DECn + 255) / 256, 256, 0, stream>>>(gbuf, cst, h_bf);
        // preds[:, t, :] = mask * (h_new @ Wfc + bfc), strided into d_out
        gemm16<0, false, 4><<<dim3((Vn / 16 + 3) / 4, 1), 128, 0, stream>>>(
            h_bf, nullptr, nullptr, WfcT, bfc, preds + (long)t * Vn, nullptr,
            Vn / 16, DECn, DECn, (long)TSTEPS * Vn, lens_sorted, t);
    }
}
